// ConceptGAE_18408229830962
// MI455X (gfx1250) — matrix operand order repaired and checked by reading, not compiled, not run on hardware
//
#include <hip/hip_runtime.h>
#include <hip/hip_bf16.h>
#include <math.h>

// ---------------------------------------------------------------------------
// Problem constants (from the reference)
// ---------------------------------------------------------------------------
constexpr int   N_  = 20000;      // nodes
constexpr int   E_  = 640000;     // edges
constexpr int   G_  = 1000;       // feature groups
constexpr int   K_  = 5;          // group width
constexpr int   H_  = 256;        // hidden dim
constexpr int   O_  = 128;        // output dim
constexpr int   GP  = 1024;       // G padded to a multiple of 32 (WMMA K-step)
constexpr int   EN  = E_ + N_;    // edges + self loops

// ---------------------------------------------------------------------------
// WMMA types (wave32, gfx1250)
// ---------------------------------------------------------------------------
typedef __attribute__((ext_vector_type(16))) __bf16 v16bf;
typedef __attribute__((ext_vector_type(8)))  __bf16 v8bf;
typedef __attribute__((ext_vector_type(8)))  float  v8f;

__device__ __forceinline__ __bf16 f2bf(float f) {
  // round-to-nearest-even float -> bf16
  unsigned u = __float_as_uint(f);
  unsigned r = (u + 0x7FFFu + ((u >> 16) & 1u)) >> 16;
  unsigned short h = (unsigned short)r;
  return *(__bf16*)&h;
}

// A fragment: 16x32 bf16, lane (0..15)=row M, half=lane>>4.
// element e -> K = ((e>>3)<<4) + half*8 + (e&7)   (two contiguous 16B chunks)
__device__ __forceinline__ v16bf load_a_frag(const __bf16* rowPtr, int half) {
  const v8bf lo = *(const v8bf*)(rowPtr + half * 8);
  const v8bf hi = *(const v8bf*)(rowPtr + 16 + half * 8);
  v16bf r;
#pragma unroll
  for (int i = 0; i < 8; ++i) { r[i] = lo[i]; r[i + 8] = hi[i]; }
  return r;
}

// B fragment: 32x16 bf16 from column-major (transposed) weights.
// lane (0..15)=col N, half=lane>>4; element e -> K = ((e>>3)<<4) + half*8 + (e&7)
__device__ __forceinline__ v16bf load_b_frag(const __bf16* colPtr, int half) {
  const v8bf lo = *(const v8bf*)(colPtr + half * 8);
  const v8bf hi = *(const v8bf*)(colPtr + 16 + half * 8);
  v16bf r;
#pragma unroll
  for (int i = 0; i < 8; ++i) { r[i] = lo[i]; r[i + 8] = hi[i]; }
  return r;
}

// ---------------------------------------------------------------------------
// GEMM: C[M x N] = A[M x K](bf16) * Bt[N x K](bf16), f32 accumulate via WMMA.
// 128 threads = 4 waves; block covers 16*MT rows x N cols; wave w owns
// NTILES = N/64 column tiles and all MT row tiles (B fragment reused MT x).
// 2-stage software pipeline over k: chunk k+1 loads are issued before the
// WMMAs of chunk k (ping-pong register sets, manual unroll-by-2).
// ---------------------------------------------------------------------------
template <int K, int N, int MT>
__global__ __launch_bounds__(128)
void wmma_gemm(const __bf16* __restrict__ A, const __bf16* __restrict__ Bt,
               float* __restrict__ C) {
  constexpr int NTILES = N / 64;
  constexpr int KC     = K / 32;           // 32 (conv1), 8 (conv2): even
  const int mBase = blockIdx.x * (16 * MT);
  const int lane  = threadIdx.x & 31;
  const int wave  = threadIdx.x >> 5;
  const int half  = lane >> 4;
  const int lidx  = lane & 15;

  const __bf16* aBase[MT];
#pragma unroll
  for (int m = 0; m < MT; ++m)
    aBase[m] = A + (size_t)(mBase + m * 16 + lidx) * K;
  const __bf16* bBase[NTILES];
#pragma unroll
  for (int t = 0; t < NTILES; ++t)
    bBase[t] = Bt + (size_t)((wave * NTILES + t) * 16 + lidx) * K;

  v8f acc[MT][NTILES];
#pragma unroll
  for (int m = 0; m < MT; ++m)
#pragma unroll
    for (int t = 0; t < NTILES; ++t)
#pragma unroll
      for (int i = 0; i < 8; ++i) acc[m][t][i] = 0.0f;

  v16bf a0[MT], b0[NTILES], a1[MT], b1[NTILES];

  // prologue: k-chunk 0
#pragma unroll
  for (int m = 0; m < MT; ++m) a0[m] = load_a_frag(aBase[m], half);
#pragma unroll
  for (int t = 0; t < NTILES; ++t) b0[t] = load_b_frag(bBase[t], half);

#pragma unroll 1
  for (int kc = 0; kc < KC; kc += 2) {
    // issue loads for chunk kc+1 (always valid: KC even)
    {
      const int ko = (kc + 1) * 32;
#pragma unroll
      for (int m = 0; m < MT; ++m) a1[m] = load_a_frag(aBase[m] + ko, half);
#pragma unroll
      for (int t = 0; t < NTILES; ++t) b1[t] = load_b_frag(bBase[t] + ko, half);
    }
    // compute chunk kc
#pragma unroll
    for (int t = 0; t < NTILES; ++t)
#pragma unroll
      for (int m = 0; m < MT; ++m)
        acc[m][t] = __builtin_amdgcn_wmma_f32_16x16x32_bf16(
            false, a0[m], false, b0[t], (short)0, acc[m][t], false, false);
    // issue loads for chunk kc+2
    if (kc + 2 < KC) {
      const int ko = (kc + 2) * 32;
#pragma unroll
      for (int m = 0; m < MT; ++m) a0[m] = load_a_frag(aBase[m] + ko, half);
#pragma unroll
      for (int t = 0; t < NTILES; ++t) b0[t] = load_b_frag(bBase[t] + ko, half);
    }
    // compute chunk kc+1
#pragma unroll
    for (int t = 0; t < NTILES; ++t)
#pragma unroll
      for (int m = 0; m < MT; ++m)
        acc[m][t] = __builtin_amdgcn_wmma_f32_16x16x32_bf16(
            false, a1[m], false, b1[t], (short)0, acc[m][t], false, false);
  }

  // C/D layout: VGPR r holds (M = r + 8*half, N = lane&15)
#pragma unroll
  for (int m = 0; m < MT; ++m)
#pragma unroll
    for (int t = 0; t < NTILES; ++t) {
      const int nBase = (wave * NTILES + t) * 16;
#pragma unroll
      for (int r = 0; r < 8; ++r) {
        const int row = mBase + m * 16 + r + half * 8;
        C[(size_t)row * N + nBase + lidx] = acc[m][t][r];
      }
    }
}

// ---------------------------------------------------------------------------
// Elementwise / graph kernels
// ---------------------------------------------------------------------------
__global__ void softmax_probs(const float* __restrict__ mfs,
                              float* __restrict__ probs) {
  int g = blockIdx.x * blockDim.x + threadIdx.x;
  if (g >= G_) return;
  float v[K_], m = -INFINITY;
#pragma unroll
  for (int k = 0; k < K_; ++k) { v[k] = mfs[g * K_ + k]; m = fmaxf(m, v[k]); }
  float s = 0.f;
#pragma unroll
  for (int k = 0; k < K_; ++k) { v[k] = expf(v[k] - m); s += v[k]; }
  float inv = 1.f / s;
#pragma unroll
  for (int k = 0; k < K_; ++k) probs[g * K_ + k] = v[k] * inv;
}

// W1[G,H] -> W1t[H, GP] bf16 (column-major for B fragments), zero-pad K
__global__ void pack_w1(const float* __restrict__ W1, __bf16* __restrict__ W1t) {
  int i = blockIdx.x * blockDim.x + threadIdx.x;   // H_*GP total
  if (i >= H_ * GP) return;
  int h = i / GP, g = i % GP;
  float v = (g < G_) ? W1[(size_t)g * H_ + h] : 0.0f;
  W1t[i] = f2bf(v);
}

// W2[H,O] -> W2t[O, H] bf16
__global__ void pack_w2(const float* __restrict__ W2, __bf16* __restrict__ W2t) {
  int i = blockIdx.x * blockDim.x + threadIdx.x;   // O_*H_ total
  if (i >= O_ * H_) return;
  int o = i / H_, h = i % H_;
  W2t[i] = f2bf(W2[(size_t)h * O_ + o]);
}

// x[N, G*K] -> x_red[N, GP] bf16 (softmax-weighted reduce over K=5), pad zeros
__global__ void reduce_x(const float* __restrict__ x,
                         const float* __restrict__ probs,
                         __bf16* __restrict__ xred) {
  long long i = (long long)blockIdx.x * blockDim.x + threadIdx.x;
  if (i >= (long long)N_ * GP) return;
  int g = (int)(i & (GP - 1));
  long long n = i >> 10;
  float v = 0.f;
  if (g < G_) {
    const float* xp = x + n * (long long)(G_ * K_) + (long long)g * K_;
    const float* pp = probs + g * K_;
#pragma unroll
    for (int k = 0; k < K_; ++k) v += xp[k] * pp[k];
  }
  xred[i] = f2bf(v);
}

__global__ void fill_f32(float* __restrict__ p, long long n) {
  long long i = (long long)blockIdx.x * blockDim.x + threadIdx.x;
  if (i < n) p[i] = 0.0f;
}

__global__ void count_deg(const int* __restrict__ dstI, float* __restrict__ deg) {
  int e = blockIdx.x * blockDim.x + threadIdx.x;
  if (e >= EN) return;
  int d = (e < E_) ? dstI[e] : (e - E_);
  atomicAdd(&deg[d], 1.0f);
}

__global__ void make_dinv(const float* __restrict__ deg, float* __restrict__ dinv) {
  int n = blockIdx.x * blockDim.x + threadIdx.x;
  if (n >= N_) return;
  float d = deg[n];
  dinv[n] = (d > 0.f) ? rsqrtf(fmaxf(d, 1e-12f)) : 0.f;
}

__global__ void make_norm(const int* __restrict__ srcI, const int* __restrict__ dstI,
                          const float* __restrict__ dinv, float* __restrict__ nrm) {
  int e = blockIdx.x * blockDim.x + threadIdx.x;
  if (e >= EN) return;
  int s, d;
  if (e < E_) { s = srcI[e]; d = dstI[e]; } else { s = d = e - E_; }
  nrm[e] = dinv[s] * dinv[d];
}

// out[dst, :] += feat[src, :] * norm ; one thread = 4 features of one edge
template <int F>
__global__ void aggregate(const float* __restrict__ feat,
                          const int* __restrict__ srcI, const int* __restrict__ dstI,
                          const float* __restrict__ nrm, float* __restrict__ out) {
  constexpr int QE = F / 4;                       // float4 quads per edge
  unsigned long long i =
      (unsigned long long)blockIdx.x * blockDim.x + threadIdx.x;
  if (i >= (unsigned long long)EN * QE) return;
  unsigned long long e = i / QE;
  int q = (int)(i % QE);
  int s, d;
  if (e < (unsigned long long)E_) { s = srcI[e]; d = dstI[e]; }
  else { s = d = (int)(e - E_); }
  float w = nrm[e];
  const float4 v = *(const float4*)(feat + (size_t)s * F + q * 4);
  float* o = out + (size_t)d * F + q * 4;
  atomicAdd(o + 0, v.x * w);
  atomicAdd(o + 1, v.y * w);
  atomicAdd(o + 2, v.z * w);
  atomicAdd(o + 3, v.w * w);
}

// h = relu(hagg + b1) -> bf16  (A operand of conv2 GEMM)
__global__ void relu_bias_bf16(const float* __restrict__ hagg,
                               const float* __restrict__ b1,
                               __bf16* __restrict__ hbf) {
  long long i = (long long)blockIdx.x * blockDim.x + threadIdx.x;
  if (i >= (long long)N_ * H_) return;
  float v = hagg[i] + b1[(int)(i & (H_ - 1))];
  hbf[i] = f2bf(fmaxf(v, 0.f));
}

// seed output with b2 so scatter-add lands on top of the bias
__global__ void init_out_bias(const float* __restrict__ b2, float* __restrict__ out) {
  long long i = (long long)blockIdx.x * blockDim.x + threadIdx.x;
  if (i >= (long long)N_ * O_) return;
  out[i] = b2[(int)(i & (O_ - 1))];
}

// ---------------------------------------------------------------------------
// Host launcher
// ---------------------------------------------------------------------------
extern "C" void kernel_launch(void* const* d_in, const int* in_sizes, int n_in,
                              void* d_out, int out_size, void* d_ws, size_t ws_size,
                              hipStream_t stream) {
  const float* x    = (const float*)d_in[0];
  const int*   eidx = (const int*)  d_in[1];   // [2, E]: first E = src, next E = dst
  const float* mfs  = (const float*)d_in[2];
  const float* W1   = (const float*)d_in[3];
  const float* b1   = (const float*)d_in[4];
  const float* W2   = (const float*)d_in[5];
  const float* b2   = (const float*)d_in[6];
  float*       out  = (float*)d_out;

  const int* srcI = eidx;
  const int* dstI = eidx + E_;

  // workspace carve-out (512B aligned blocks)
  size_t off = 0;
  auto alloc = [&](size_t bytes) -> char* {
    char* p = (char*)d_ws + off;
    off += (bytes + 511) & ~(size_t)511;
    return p;
  };
  float*  probs = (float*) alloc((size_t)G_ * K_ * 4);
  float*  deg   = (float*) alloc((size_t)N_ * 4);
  float*  dinv  = (float*) alloc((size_t)N_ * 4);
  float*  nrm   = (float*) alloc((size_t)EN * 4);
  __bf16* W1t   = (__bf16*)alloc((size_t)H_ * GP * 2);
  __bf16* W2t   = (__bf16*)alloc((size_t)O_ * H_ * 2);
  __bf16* xred  = (__bf16*)alloc((size_t)N_ * GP * 2);
  float*  h0    = (float*) alloc((size_t)N_ * H_ * 4);
  float*  hagg  = (float*) alloc((size_t)N_ * H_ * 4);
  __bf16* hbf   = (__bf16*)alloc((size_t)N_ * H_ * 2);
  float*  z0    = h0;  // h0 dead after relu_bias_bf16 -> reuse for conv2 GEMM out
  (void)ws_size; (void)in_sizes; (void)n_in; (void)out_size;

  const int T = 256;
  auto blk = [](long long n, int t) { return (unsigned)((n + t - 1) / t); };

  // --- prep: softmax probs, bf16 transposed weights ---
  softmax_probs<<<blk(G_, T), T, 0, stream>>>(mfs, probs);
  pack_w1<<<blk((long long)H_ * GP, T), T, 0, stream>>>(W1, W1t);
  pack_w2<<<blk((long long)O_ * H_, T), T, 0, stream>>>(W2, W2t);

  // --- x -> x_red (bf16, K padded to 1024) : the 400MB streaming pass ---
  reduce_x<<<blk((long long)N_ * GP, T), T, 0, stream>>>(x, probs, xred);

  // --- degrees / symmetric norm (self-loops are edges E..E+N-1) ---
  fill_f32<<<blk(N_, T), T, 0, stream>>>(deg, N_);
  count_deg<<<blk(EN, T), T, 0, stream>>>(dstI, deg);
  make_dinv<<<blk(N_, T), T, 0, stream>>>(deg, dinv);
  make_norm<<<blk(EN, T), T, 0, stream>>>(srcI, dstI, dinv, nrm);

  // --- conv1: h0 = x_red @ W1  (WMMA bf16 -> f32), then scatter, relu+b1 ---
  wmma_gemm<GP, H_, 2><<<N_ / 32, 128, 0, stream>>>(xred, W1t, h0);
  fill_f32<<<blk((long long)N_ * H_, T), T, 0, stream>>>(hagg, (long long)N_ * H_);
  aggregate<H_><<<blk((long long)EN * (H_ / 4), T), T, 0, stream>>>(
      h0, srcI, dstI, nrm, hagg);
  relu_bias_bf16<<<blk((long long)N_ * H_, T), T, 0, stream>>>(hagg, b1, hbf);

  // --- conv2: z0 = h @ W2 (WMMA), then scatter onto b2-seeded output ---
  wmma_gemm<H_, O_, 2><<<N_ / 32, 128, 0, stream>>>(hbf, W2t, z0);
  init_out_bias<<<blk((long long)N_ * O_, T), T, 0, stream>>>(b2, out);
  aggregate<O_><<<blk((long long)EN * (O_ / 4), T), T, 0, stream>>>(
      z0, srcI, dstI, nrm, out);
}
